// Attention_12927851561046
// MI455X (gfx1250) — compile-verified, matrix operand order checked by reading, not run on hardware
//
#include <hip/hip_runtime.h>
#include <hip/hip_bf16.h>

typedef __attribute__((ext_vector_type(16))) _Float16 v16h;
typedef __attribute__((ext_vector_type(4)))  _Float16 v4h;
typedef __attribute__((ext_vector_type(8)))  float    v8f;

#define N_TOK   343
#define NP      352
#define NT      22
#define NPAD    344     // bias row stride (floats): 344*4 B = 16B aligned rows
#define DMODEL  128
#define DH      32
#define NHEADS  4
#define VT_LD   360     // vT row stride (halfs): 720 B, 16B aligned, bank step 52
#define WSL_LD  136     // wslT row stride (halfs): 272 B, 16B aligned, bank step 4
#define WOL_LD  134     // wolT row stride (halfs): 268 B, 4B aligned
#define QK_SCALE 0.17677669529663689f   // 32^-0.5

// ---- WMMA fragment helper (layouts per CDNA5 ISA 7.12.2, wave32) ----
// A (16x32 f16): lane%16 = M row;  B (32x16 f16): lane%16 = N col.
// K position of VGPR v, halfword h: 2v + (v>=4 ? 8:0) + (lane>=16 ? 8:0) + h
// All operands are stored K-contiguous in LDS, so each fragment lowers to
// two 16B ds_load_b128 (when the row base is 16B aligned).
__device__ __forceinline__ v16h frag_k_contig(const _Float16* rowptr, int lane) {
    v16h f;
#pragma unroll
    for (int v = 0; v < 8; ++v) {
        int k = 2 * v + ((v & 4) ? 8 : 0) + ((lane & 16) ? 8 : 0);
        f[2 * v]     = rowptr[k];
        f[2 * v + 1] = rowptr[k + 1];
    }
    return f;
}

__device__ __forceinline__ v8f zero8() {
    v8f z;
#pragma unroll
    for (int i = 0; i < 8; ++i) z[i] = 0.0f;
    return z;
}

__device__ __forceinline__ v8f wmma_f16(v16h a, v16h b, v8f c) {
    return __builtin_amdgcn_wmma_f32_16x16x32_f16(false, a, false, b, (short)0, c,
                                                  false, false);
}

// ---- kernel 0: bias_t[h][j][i(padded 344)] = bias_table[rel_idx[i,j], h] ----
// Transposed + padded so the attention kernel can fetch 8 consecutive i's per
// lane as two aligned float4 loads.
__global__ void bias_gather_kernel(const float* __restrict__ bias_table,
                                   const int*   __restrict__ rel_idx,
                                   float*       __restrict__ bias_ws) {
    int t = blockIdx.x * blockDim.x + threadIdx.x;
    const int total = NHEADS * N_TOK * NPAD;
    if (t >= total) return;
    int i  = t % NPAD;
    int hj = t / NPAD;
    int j  = hj % N_TOK;
    int h  = hj / N_TOK;
    bias_ws[t] = (i < N_TOK)
        ? bias_table[(size_t)rel_idx[(size_t)i * N_TOK + j] * NHEADS + h]
        : 0.0f;
}

// ---- kernel 1: fully fused window attention, one workgroup per window ----
__global__ __launch_bounds__(256, 1) void win_attn_kernel(
    const float* __restrict__ xg,
    const float* __restrict__ w_qkv,
    const float* __restrict__ w_out,
    const float* __restrict__ bias_ws,
    float*       __restrict__ out) {
    extern __shared__ _Float16 smem[];
    _Float16* xf   = smem;                  // 352*128 = 45056 halfs
    _Float16* qh   = xf + NP * DMODEL;      // 352*32 per-head Q (pre-scaled)
    _Float16* kh   = qh + NP * DH;          // 352*32 per-head K
    _Float16* vT   = kh + NP * DH;          // 32 x 360: V transposed (K-contig)
    _Float16* oacc = vT + DH * VT_LD;       // 352*128 attention output
    _Float16* wslT = oacc + NP * DMODEL;    // 96 x 136: qkv slice transposed
    _Float16* pbuf = wslT + 96 * WSL_LD;    // 8 waves * 16*32 P bounce slots
    _Float16* wolT = wslT;                  // 128 x 134 = 17152 halfs, aliases
                                            // wslT(13056)+pbuf(4096) exactly
    // total: 141312 halfs = 282624 bytes (< 320 KB WGP LDS)

    const int tid  = threadIdx.x;
    const int lane = tid & 31;
    const int wave = tid >> 5;
    const int win  = blockIdx.x;
    const int lm   = lane & 15;             // N/M index within a 16-wide tile
    const int hi16 = (lane & 16) ? 8 : 0;   // C-layout row offset, upper half

    // stage x window (343x128 f32 -> f16, zero-pad rows to 352), float4 reads
    const float4* xw4 = (const float4*)(xg + (size_t)win * N_TOK * DMODEL);
    for (int i4 = tid; i4 < NP * DMODEL / 4; i4 += 256) {
        int r = (i4 * 4) >> 7;
        v4h hv;
        if (r < N_TOK) {
            float4 f = xw4[i4];
            hv[0] = (_Float16)f.x; hv[1] = (_Float16)f.y;
            hv[2] = (_Float16)f.z; hv[3] = (_Float16)f.w;
        } else {
            hv[0] = hv[1] = hv[2] = hv[3] = (_Float16)0.0f;
        }
        *(v4h*)(&xf[i4 * 4]) = hv;
    }

    for (int h = 0; h < NHEADS; ++h) {
        // stage this head's 128x96 slice of w_qkv, transposed (K-contiguous)
        for (int i = tid; i < DMODEL * 96; i += 256) {
            int r = i / 96, c = i % 96;
            int col = (c >> 5) * DMODEL + h * DH + (c & 31);
            wslT[c * WSL_LD + r] = (_Float16)w_qkv[(size_t)r * (3 * DMODEL) + col];
        }
        __syncthreads();  // also guards prev head's phase-2 reads of qh/kh/vT

        // ---- phase 1: QKV projection, split by destination (uniform stores) --
        // Q tiles -> qh (pre-scaled)
        for (int jt = wave; jt < NT * 2; jt += 8) {
            int rt = jt >> 1, c0 = (jt & 1) * 16;
            v8f acc = zero8();
#pragma unroll
            for (int kc = 0; kc < 4; ++kc) {
                v16h a = frag_k_contig(&xf[(rt * 16 + lm) * DMODEL + kc * 32], lane);
                v16h b = frag_k_contig(&wslT[(c0 + lm) * WSL_LD + kc * 32], lane);
                acc = wmma_f16(a, b, acc);
            }
#pragma unroll
            for (int v = 0; v < 8; ++v)
                qh[(rt * 16 + v + hi16) * DH + c0 + lm] =
                    (_Float16)(acc[v] * QK_SCALE);
        }
        // K tiles -> kh
        for (int jt = wave; jt < NT * 2; jt += 8) {
            int rt = jt >> 1, c0 = (jt & 1) * 16;
            v8f acc = zero8();
#pragma unroll
            for (int kc = 0; kc < 4; ++kc) {
                v16h a = frag_k_contig(&xf[(rt * 16 + lm) * DMODEL + kc * 32], lane);
                v16h b = frag_k_contig(&wslT[(32 + c0 + lm) * WSL_LD + kc * 32], lane);
                acc = wmma_f16(a, b, acc);
            }
#pragma unroll
            for (int v = 0; v < 8; ++v)
                kh[(rt * 16 + v + hi16) * DH + c0 + lm] = (_Float16)acc[v];
        }
        // V tiles -> vT (transposed; each lane's 8 rows are contiguous)
        for (int jt = wave; jt < NT * 2; jt += 8) {
            int rt = jt >> 1, c0 = (jt & 1) * 16;
            v8f acc = zero8();
#pragma unroll
            for (int kc = 0; kc < 4; ++kc) {
                v16h a = frag_k_contig(&xf[(rt * 16 + lm) * DMODEL + kc * 32], lane);
                v16h b = frag_k_contig(&wslT[(64 + c0 + lm) * WSL_LD + kc * 32], lane);
                acc = wmma_f16(a, b, acc);
            }
#pragma unroll
            for (int v = 0; v < 8; ++v)
                vT[(c0 + lm) * VT_LD + rt * 16 + hi16 + v] = (_Float16)acc[v];
        }
        __syncthreads();

        // ---- phase 2: attention; each wave owns query tiles qt, qt+8, ... ----
        const float* bh = bias_ws + (size_t)h * N_TOK * NPAD;
        for (int qt = wave; qt < NT; qt += 8) {
            // S row-block (16 x 352) in registers; dh=32 => one WMMA per tile
            v16h qf = frag_k_contig(&qh[(qt * 16 + lm) * DH], lane);
            v8f s[NT];
#pragma unroll
            for (int nt = 0; nt < NT; ++nt) {
                v16h kf = frag_k_contig(&kh[(nt * 16 + lm) * DH], lane);
                s[nt] = wmma_f16(qf, kf, zero8());
            }
            // + relative-position bias (two aligned float4 per tile), mask pads
            int ib = qt * 16 + hi16;          // this lane's first row
            if (ib > 336) ib = 336;           // clamp: rows >=343 are garbage
#pragma unroll
            for (int nt = 0; nt < NT; ++nt) {
                int j = nt * 16 + lm;
                bool jv = (j < N_TOK);
                float4 b0, b1;
                b0.x = b0.y = b0.z = b0.w = 0.0f;
                b1 = b0;
                if (jv) {
                    const float4* bp =
                        (const float4*)(bh + (size_t)j * NPAD + ib);
                    b0 = bp[0];
                    b1 = bp[1];
                }
                float bb[8] = {b0.x, b0.y, b0.z, b0.w, b1.x, b1.y, b1.z, b1.w};
#pragma unroll
                for (int v = 0; v < 8; ++v)
                    s[nt][v] = jv ? (s[nt][v] + bb[v]) : -__builtin_inff();
            }
            // softmax over 352 cols: in-lane partials + 16-lane xor reductions
            float rmax[8], rsum[8];
#pragma unroll
            for (int v = 0; v < 8; ++v) rmax[v] = -__builtin_inff();
#pragma unroll
            for (int nt = 0; nt < NT; ++nt)
#pragma unroll
                for (int v = 0; v < 8; ++v) rmax[v] = fmaxf(rmax[v], s[nt][v]);
#pragma unroll
            for (int v = 0; v < 8; ++v) {
#pragma unroll
                for (int m = 1; m < 16; m <<= 1)
                    rmax[v] = fmaxf(rmax[v], __shfl_xor(rmax[v], m, 32));
                rsum[v] = 0.0f;
            }
#pragma unroll
            for (int nt = 0; nt < NT; ++nt)
#pragma unroll
                for (int v = 0; v < 8; ++v) {
                    float p = __expf(s[nt][v] - rmax[v]);
                    s[nt][v] = p;
                    rsum[v] += p;
                }
#pragma unroll
            for (int v = 0; v < 8; ++v) {
#pragma unroll
                for (int m = 1; m < 16; m <<= 1)
                    rsum[v] += __shfl_xor(rsum[v], m, 32);
                rsum[v] = 1.0f / rsum[v];
            }
#pragma unroll
            for (int nt = 0; nt < NT; ++nt)
#pragma unroll
                for (int v = 0; v < 8; ++v) s[nt][v] *= rsum[v];

            // ---- O = P @ V : 11 chunks of K=32; C-layout -> A-layout via LDS ----
            _Float16* pw = pbuf + wave * 16 * 32;  // per-wave private slot
            v8f o0 = zero8(), o1 = zero8();
#pragma unroll
            for (int c = 0; c < 11; ++c) {
#pragma unroll
                for (int v = 0; v < 8; ++v) {
                    int m = v + hi16;
                    pw[m * 32 + lm]      = (_Float16)s[2 * c][v];
                    pw[m * 32 + 16 + lm] = (_Float16)s[2 * c + 1][v];
                }
                asm volatile("s_wait_dscnt 0x0" ::: "memory");  // wave LDS RAW fence
                v16h pf  = frag_k_contig(&pw[lm * 32], lane);
                v16h vf0 = frag_k_contig(&vT[lm * VT_LD + c * 32], lane);
                v16h vf1 = frag_k_contig(&vT[(16 + lm) * VT_LD + c * 32], lane);
                o0 = wmma_f16(pf, vf0, o0);
                o1 = wmma_f16(pf, vf1, o1);
                asm volatile("s_wait_dscnt 0x0" ::: "memory");  // WAR vs next chunk
            }
#pragma unroll
            for (int v = 0; v < 8; ++v) {
                int row = qt * 16 + v + hi16;
                oacc[row * DMODEL + h * DH + lm]      = (_Float16)o0[v];
                oacc[row * DMODEL + h * DH + 16 + lm] = (_Float16)o1[v];
            }
        }
        __syncthreads();
    }

    // ---- final projection: out = oacc(352x128) @ w_out(128x128) ----
    for (int i = tid; i < DMODEL * DMODEL; i += 256) {
        int k = i >> 7, n = i & 127;
        wolT[n * WOL_LD + k] = (_Float16)w_out[i];   // transposed, K-contiguous
    }
    __syncthreads();

    for (int qt = wave; qt < NT; qt += 8) {
        for (int ct = 0; ct < 8; ++ct) {
            v8f acc = zero8();
#pragma unroll
            for (int kc = 0; kc < 4; ++kc) {
                v16h a = frag_k_contig(&oacc[(qt * 16 + lm) * DMODEL + kc * 32], lane);
                v16h b = frag_k_contig(&wolT[(ct * 16 + lm) * WOL_LD + kc * 32], lane);
                acc = wmma_f16(a, b, acc);
            }
#pragma unroll
            for (int v = 0; v < 8; ++v) {
                int row = qt * 16 + v + hi16;
                if (row < N_TOK)
                    out[((size_t)win * N_TOK + row) * DMODEL + ct * 16 + lm] = acc[v];
            }
        }
    }
}

extern "C" void kernel_launch(void* const* d_in, const int* in_sizes, int n_in,
                              void* d_out, int out_size, void* d_ws, size_t ws_size,
                              hipStream_t stream) {
    const float* x        = (const float*)d_in[0];
    const float* w_qkv    = (const float*)d_in[1];
    const float* w_out    = (const float*)d_in[2];
    const float* bias_tab = (const float*)d_in[3];
    const int*   rel_idx  = (const int*)d_in[4];
    float* out     = (float*)d_out;
    float* bias_ws = (float*)d_ws;  // 4*343*344 f32 = 1.89 MB (L2-resident)

    const int total = NHEADS * N_TOK * NPAD;
    bias_gather_kernel<<<(total + 255) / 256, 256, 0, stream>>>(bias_tab, rel_idx,
                                                                bias_ws);
    // 512 windows, 8 waves each, 282624 B dynamic LDS per workgroup
    win_attn_kernel<<<512, 256, 282624, stream>>>(x, w_qkv, w_out, bias_ws, out);
}
